// Model_44152263803051
// MI455X (gfx1250) — matrix operand order repaired and checked by reading, not compiled, not run on hardware
//
#include <hip/hip_runtime.h>
#include <hip/hip_bf16.h>
#include <stdint.h>

// ---------------------------------------------------------------------------
// Fully fused per-batch pipeline for MI455X (gfx1250, wave32, WMMA).
//   grid = 4096 workgroups (one per batch item), 256 threads (8 waves).
//   Phase 1: load idx, L2-normalize, f16 -> LDS; zero padded images
//   Phase 2: Gram C = f f^T via v_wmma_f32_16x16x32_f16 (2x2 tiles, K=768),
//            all-b128 fragment loads; bins -> LDS; uniform embed scatter
//            into CHANNEL-LAST padded image [24][20][32] (C padded 20->32)
//   Phase 3: conv1 (20->32, 3x3, pad2) implicit-GEMM WMMA with K ordered as
//            (ky,kx,ic): B-fragments are contiguous v16h at compile-time
//            pixel offsets -> pure b128 DS pipeline + ReLU + maxpool to
//            channel-last padded pool1 [15][13][32]
//   Phase 4: conv2 (32->64, 3x3, pad2) same scheme + ReLU + maxpool
//   Phase 5: FC 1920->3, deterministic two-level reduction
// All intermediates stay in LDS; only idx (264 MB total) streams from HBM
// (roofline floor ~11 us at 23.3 TB/s).  92.4 KB LDS -> 3 workgroups/WGP.
// ---------------------------------------------------------------------------

typedef _Float16 half_t;
typedef __attribute__((ext_vector_type(16))) _Float16 v16h;
typedef __attribute__((ext_vector_type(8)))  _Float16 v8h;
typedef __attribute__((ext_vector_type(8)))  float    v8f;

#define WG 256

// LDS region layout (bytes), reused across phases (barriers separate them):
#define OFF_F    0       // phase1-2: f16 f[32][768]                 = 49152
#define OFF_EP   49152   // phase2-3: f16 e_cl[24][20][32]           = 30720 (end 79872)
#define OFF_P1   79872   // phase3-4: f16 p1_cl[15][13][32]          = 12480 (end 92352)
#define OFF_W1   0       // phase3:   f16 w1[32][9*32]               = 18432
#define OFF_C1   18432   // phase3:   f16 c1out[32][22][18]          = 25344 (end 43776)
#define OFF_W2   0       // phase4:   f16 w2[64][9*32]               = 36864
#define OFF_C2   36864   // phase4:   f16 c2out[64][13][11]          = 18304 (end 55168)
#define OFF_P2   55168   // phase4-5: f32 pool2[1920]                =  7680 (end 62848)
#define SMEM_BYTES 92352

// ---- WMMA fragment loaders (contiguous k = g*16 + i for A and B, wave32) ------
// A-matrix 16x32 f16 from a row-major row pointer (row = lane&15, at +k0):
//   a[0..7]  = row[k0 + g*16 + 0..7]       (contiguous 16B  -> ds_load_b128)
//   a[8..15] = row[k0 + g*16 + 8..15]      (contiguous 16B  -> ds_load_b128)
__device__ __forceinline__ v16h loadA16(const half_t* base, int g) {
  v8h lo = *(const v8h*)(base + g * 16);
  v8h hi = *(const v8h*)(base + g * 16 + 8);
  return __builtin_shufflevector(lo, hi,
                                 0, 1, 2, 3, 4, 5, 6, 7,
                                 8, 9, 10, 11, 12, 13, 14, 15);
}
// B-matrix 32x16 f16, col = lane&15: bm[i] = B[k0 + g*16 + i][n].
// With contiguous-K storage this is one 32B v16h (2x ds_load_b128).
__device__ __forceinline__ v16h loadB16(const half_t* base, int g) {
  return *(const v16h*)(base + g * 16);
}

__device__ __forceinline__ v8f wmma_f16(v16h a, v16h b, v8f c) {
  // D = A*B + C : v_wmma_f32_16x16x32_f16
  return __builtin_amdgcn_wmma_f32_16x16x32_f16(
      /*neg_a=*/false, a, /*neg_b=*/false, b,
      /*c_mod=*/(short)0, c, /*reuse_a=*/false, /*reuse_b=*/false);
}

__global__ __launch_bounds__(WG) void fused_pipeline_kernel(
    const float* __restrict__ idx,   // (4096, 21, 768)
    const float* __restrict__ cosE,  // (12, 16)
    const float* __restrict__ w1g,   // (32, 20, 3, 3)
    const float* __restrict__ b1g,   // (32,)
    const float* __restrict__ w2g,   // (64, 32, 3, 3)
    const float* __restrict__ b2g,   // (64,)
    const float* __restrict__ owg,   // (3, 1920)
    const float* __restrict__ obg,   // (3,)
    float* __restrict__ out)         // (4096, 3)
{
  extern __shared__ unsigned char smem[];
  __shared__ half_t cosL[12 * 16];
  __shared__ int    binsL[400];
  __shared__ float  b1L[32];
  __shared__ float  b2L[64];
  __shared__ float  fcPart[8][3];

  half_t* fBuf = (half_t*)(smem + OFF_F);
  half_t* eCL  = (half_t*)(smem + OFF_EP);   // [24][20][32] channel-last
  half_t* p1P  = (half_t*)(smem + OFF_P1);   // [15][13][32] channel-last
  half_t* w1B  = (half_t*)(smem + OFF_W1);   // [32][9*32]  k=(p,ic)
  half_t* c1o  = (half_t*)(smem + OFF_C1);   // [32][22][18]
  half_t* w2B  = (half_t*)(smem + OFF_W2);   // [64][9*32]  k=(p,ic)
  half_t* c2o  = (half_t*)(smem + OFF_C2);   // [64][13][11]
  float*  p2B  = (float*)(smem + OFF_P2);

  const int tid  = threadIdx.x;
  const int lane = tid & 31;          // wave32
  const int wave = tid >> 5;          // 8 waves
  const int lr   = lane & 15;
  const int g    = lane >> 4;
  const int b    = blockIdx.x;

  // ---------------- Phase 1: preload + normalize idx rows -> f16 LDS -------
  const float* idxb = idx + (size_t)b * (21 * 768);
  for (int i = tid; i < 192; i += WG) cosL[i] = (half_t)cosE[i];
  if (tid < 32) b1L[tid] = b1g[tid];
  if (tid >= 32 && tid < 96) b2L[tid - 32] = b2g[tid - 32];
  for (int i = tid; i < 11 * 768; i += WG) fBuf[21 * 768 + i] = (half_t)0.f;   // pad rows
  for (int i = tid; i < 24 * 20 * 32; i += WG) eCL[i] = (half_t)0.f;           // halo+Cpad
  for (int i = tid; i < 15 * 13 * 32; i += WG) p1P[i] = (half_t)0.f;           // halo

  for (int row = wave; row < 21; row += 8) {
    const float* rp = idxb + row * 768;
    float vals[24];
    float ss = 0.f;
#pragma unroll
    for (int i = 0; i < 24; ++i) { float v = rp[lane + 32 * i]; vals[i] = v; ss += v * v; }
#pragma unroll
    for (int off = 16; off > 0; off >>= 1) ss += __shfl_xor(ss, off, 32);
    float inv = 1.f / fmaxf(sqrtf(ss), 1e-12f);
#pragma unroll
    for (int i = 0; i < 24; ++i) fBuf[row * 768 + lane + 32 * i] = (half_t)(vals[i] * inv);
  }
  __syncthreads();

  // ---------------- Phase 2: Gram via WMMA -> bins -> embed scatter --------
  if (wave < 4) {                      // one 16x16 tile per wave (2x2 tiles)
    int mT = wave >> 1, nT = wave & 1;
    v8f acc = {};
    const half_t* Abase = fBuf + (mT * 16 + lr) * 768;
    const half_t* Bbase = fBuf + (nT * 16 + lr) * 768;
    for (int s = 0; s < 24; ++s) {     // K = 768 = 24 * 32
      v16h a  = loadA16(Abase + 32 * s, g);
      v16h bm = loadB16(Bbase + 32 * s, g);
      acc = wmma_f16(a, bm, acc);
    }
#pragma unroll
    for (int r = 0; r < 8; ++r) {      // C layout: m = r + 8*g, n = lane&15
      int m = mT * 16 + r + 8 * g;     // channel index
      int n = nT * 16 + lr;            // spatial H index
      if (m < 20 && n < 20) {
        float c = acc[r];
        binsL[m * 20 + n] =
            (c >= 1.f) ? 11 : ((c < 0.1f) ? 1 : ((int)floorf(c * 10.f) + 1));
      }
    }
  }
  __syncthreads();

  // uniform (convergent) embedding scatter: image[ic=m][h=n][w=t], pad 2,
  // channel-last: eCL[((h+2)*20 + (w+2))*32 + ic]
  for (int i = tid; i < 20 * 20 * 16; i += WG) {
    int m = i / 320, rem = i - m * 320;
    int n = rem >> 4, t = rem & 15;
    eCL[((n + 2) * 20 + (t + 2)) * 32 + m] = cosL[binsL[m * 20 + n] * 16 + t];
  }
  __syncthreads();

  // ---------------- Phase 3: conv1 implicit-GEMM WMMA ----------------------
  // M=32 oc (2 tiles), N=396 pixels (25 tiles), K=9*32 (9 steps, k=(p,ic))
  for (int i = tid; i < 32 * 288; i += WG) {
    int oc = i / 288, rem = i - oc * 288;
    int p = rem >> 5, ic = rem & 31;
    w1B[i] = (ic < 20) ? (half_t)w1g[oc * 180 + ic * 9 + p] : (half_t)0.f;
  }
  __syncthreads();

  for (int tile = wave; tile < 50; tile += 8) {
    int mT = tile / 25, nT = tile - mT * 25;
    int n = nT * 16 + lr;
    bool nv = n < 396;
    int y = n / 18, x = n - y * 18;
    const half_t* eBase = eCL + (nv ? ((y * 20 + x) * 32) : 0) + g * 16;
    v8f acc = {};
    const half_t* Abase = w1B + (mT * 16 + lr) * 288;
#pragma unroll
    for (int p = 0; p < 9; ++p) {      // filter positions; const pixel offsets
      int ky = p / 3, kx = p - ky * 3;
      v16h a  = loadA16(Abase + 32 * p, g);
      v16h bm = *(const v16h*)(eBase + (ky * 20 + kx) * 32);
      acc = wmma_f16(a, bm, acc);
    }
    if (nv) {
#pragma unroll
      for (int r = 0; r < 8; ++r) {
        int oc = mT * 16 + r + 8 * g;
        float o = fmaxf(acc[r] + b1L[oc], 0.f);
        c1o[(oc * 22 + y) * 18 + x] = (half_t)o;
      }
    }
  }
  __syncthreads();

  // maxpool1: 32 x 22x18 -> channel-last padded [15][13][32], interior at +2
  for (int i = tid; i < 32 * 11 * 9; i += WG) {
    int oc = i / 99, rem = i - oc * 99;
    int py = rem / 9, px = rem - py * 9;
    const half_t* src = c1o + (oc * 22 + 2 * py) * 18 + 2 * px;
    float m0 = fmaxf((float)src[0],  (float)src[1]);
    float m1 = fmaxf((float)src[18], (float)src[19]);
    p1P[((py + 2) * 13 + (px + 2)) * 32 + oc] = (half_t)fmaxf(m0, m1);
  }
  __syncthreads();

  // ---------------- Phase 4: conv2 implicit-GEMM WMMA ----------------------
  // M=64 oc (4 tiles), N=143 pixels (9 tiles), K=9*32 (9 steps, k=(p,ic))
  for (int i = tid; i < 64 * 288; i += WG) {
    int oc = i / 288, rem = i - oc * 288;
    int p = rem >> 5, ic = rem & 31;
    w2B[i] = (half_t)w2g[oc * 288 + ic * 9 + p];
  }
  __syncthreads();

  for (int tile = wave; tile < 36; tile += 8) {
    int mT = tile / 9, nT = tile - mT * 9;
    int n = nT * 16 + lr;
    bool nv = n < 143;
    int y = n / 11, x = n - y * 11;
    const half_t* pBase = p1P + (nv ? ((y * 13 + x) * 32) : 0) + g * 16;
    v8f acc = {};
    const half_t* Abase = w2B + (mT * 16 + lr) * 288;
#pragma unroll
    for (int p = 0; p < 9; ++p) {
      int ky = p / 3, kx = p - ky * 3;
      v16h a  = loadA16(Abase + 32 * p, g);
      v16h bm = *(const v16h*)(pBase + (ky * 13 + kx) * 32);
      acc = wmma_f16(a, bm, acc);
    }
    if (nv) {
#pragma unroll
      for (int r = 0; r < 8; ++r) {
        int oc = mT * 16 + r + 8 * g;
        float o = fmaxf(acc[r] + b2L[oc], 0.f);
        c2o[(oc * 13 + y) * 11 + x] = (half_t)o;
      }
    }
  }
  __syncthreads();

  // maxpool2: 64 x 13x11 -> 64 x 6x5 (flatten order oc*30 + py*5 + px)
  for (int i = tid; i < 1920; i += WG) {
    int oc = i / 30, rem = i - oc * 30;
    int py = rem / 5, px = rem - py * 5;
    const half_t* src = c2o + (oc * 13 + 2 * py) * 11 + 2 * px;
    float m0 = fmaxf((float)src[0],  (float)src[1]);
    float m1 = fmaxf((float)src[11], (float)src[12]);
    p2B[i] = fmaxf(m0, m1);
  }
  __syncthreads();

  // ---------------- Phase 5: FC 1920 -> 3 (deterministic reduce) -----------
  float a0 = 0.f, a1 = 0.f, a2 = 0.f;
  for (int j = tid; j < 1920; j += WG) {
    float v = p2B[j];
    a0 += v * owg[j];
    a1 += v * owg[1920 + j];
    a2 += v * owg[3840 + j];
  }
#pragma unroll
  for (int off = 16; off > 0; off >>= 1) {
    a0 += __shfl_xor(a0, off, 32);
    a1 += __shfl_xor(a1, off, 32);
    a2 += __shfl_xor(a2, off, 32);
  }
  if (lane == 0) { fcPart[wave][0] = a0; fcPart[wave][1] = a1; fcPart[wave][2] = a2; }
  __syncthreads();
  if (tid < 3) {
    float s = obg[tid];
#pragma unroll
    for (int w = 0; w < 8; ++w) s += fcPart[w][tid];
    out[b * 3 + tid] = s;
  }
}

extern "C" void kernel_launch(void* const* d_in, const int* in_sizes, int n_in,
                              void* d_out, int out_size, void* d_ws, size_t ws_size,
                              hipStream_t stream) {
  (void)in_sizes; (void)n_in; (void)out_size; (void)d_ws; (void)ws_size;
  const float* idx  = (const float*)d_in[0];
  // d_in[1..3] = idx1..idx3 (unused by reference math)
  const float* cosE = (const float*)d_in[4];
  const float* w1   = (const float*)d_in[5];
  const float* b1   = (const float*)d_in[6];
  const float* w2   = (const float*)d_in[7];
  const float* b2   = (const float*)d_in[8];
  const float* ow   = (const float*)d_in[9];
  const float* ob   = (const float*)d_in[10];
  float* out = (float*)d_out;

  fused_pipeline_kernel<<<dim3(4096), dim3(WG), SMEM_BYTES, stream>>>(
      idx, cosE, w1, b1, w2, b2, ow, ob, out);
}
